// Torch_fwRF_voxel_block_25572235280403
// MI455X (gfx1250) — matrix operand order repaired and checked by reading, not compiled, not run on hardware
//
#include <hip/hip_runtime.h>
#include <hip/hip_bf16.h>
#include <cstdint>

// ---------------------------------------------------------------------------
// B=128, V=512, REZ=[56,28,14,7], CH=[64,128,256,512], C=960.
// Stage 1: 4 GEMMs (B*Ck, n^2) x (n^2, V) via bf16 WMMA (f32 accumulate).
// Stage 2 folded into the GEMM epilogue:
//   w2[c,v]     = weights[v,c] / features_s[c,v]
//   bias_eff[v] = bias[v] - sum_c features_m[c,v]*w2[c,v]
//   r[b,v]      = sum_c pooled[b,c,v]*w2[c,v] + bias_eff[v]
// Deterministic: per-tile partials in workspace + final reduction (no atomics).
// ---------------------------------------------------------------------------

typedef __attribute__((ext_vector_type(16))) __bf16 v16bf_t;
typedef __attribute__((ext_vector_type(8)))  float  v8f_t;

#define NV 512       // voxels
#define NB 128       // batch
#define NC 960       // total channels
#define NTILES 7680  // total 16-row M tiles across all 4 resolutions

__device__ __forceinline__ unsigned short f2bfu(float f) {
    __bf16 h = (__bf16)f;
    return __builtin_bit_cast(unsigned short, h);
}

// ---------------- prep kernels -------------------------------------------

__global__ void prep_w2_kernel(const float* __restrict__ weights,  // (V, C)
                               const float* __restrict__ fs,       // (C, V)
                               float* __restrict__ w2)             // (C, V)
{
    int idx = blockIdx.x * blockDim.x + threadIdx.x;
    if (idx >= NC * NV) return;
    int c = idx / NV;
    int v = idx - c * NV;
    w2[idx] = weights[(size_t)v * NC + c] / fs[idx];
}

__global__ void prep_bias_kernel(const float* __restrict__ bias,   // (V)
                                 const float* __restrict__ fm,     // (C, V)
                                 const float* __restrict__ w2,     // (C, V)
                                 float* __restrict__ bias_eff)     // (V)
{
    int v = blockIdx.x * blockDim.x + threadIdx.x;
    if (v >= NV) return;
    float s = bias[v];
    for (int c = 0; c < NC; ++c)
        s -= fm[(size_t)c * NV + v] * w2[(size_t)c * NV + v];
    bias_eff[v] = s;
}

// Convert prf (V, K) f32 -> (V, Kpad) bf16, zero padded in K.
__global__ void prf_to_bf16_kernel(const float* __restrict__ prf,
                                   unsigned short* __restrict__ dst,
                                   int K, int Kpad)
{
    int idx = blockIdx.x * blockDim.x + threadIdx.x;
    int total = NV * Kpad;
    if (idx >= total) return;
    int v = idx / Kpad;
    int k = idx - v * Kpad;
    float val = (k < K) ? prf[(size_t)v * K + k] : 0.0f;
    dst[idx] = f2bfu(val);
}

// ---------------- main fused GEMM kernel ----------------------------------

// Unchecked vector load of 8 consecutive f32 (16B-aligned rows only).
__device__ __forceinline__ void load8v(const float* __restrict__ row,
                                       int k0, float* o)
{
    float4 x = *reinterpret_cast<const float4*>(row + k0);
    float4 y = *reinterpret_cast<const float4*>(row + k0 + 4);
    o[0] = x.x; o[1] = x.y; o[2] = x.z; o[3] = x.w;
    o[4] = y.x; o[5] = y.y; o[6] = y.z; o[7] = y.w;
}

// Branchless bounded load: clamp index (always-safe load) + select to zero.
// No EXEC manipulation -> plain v_cndmask codegen, no divergent branches.
__device__ __forceinline__ void load8t(const float* __restrict__ row,
                                       int k0, int K, float* o)
{
#pragma unroll
    for (int i = 0; i < 8; ++i) {
        int k = k0 + i;
        bool in = (k < K);
        float x = row[in ? k : 0];
        o[i] = in ? x : 0.0f;
    }
}

// One workgroup = one 16-row M tile (one b, 16 consecutive c), all N=512.
// 8 waves; wave w owns N columns [w*64, w*64+64) as four 16x16 WMMA tiles.
__global__ void __launch_bounds__(256)
pooled_gemm_kernel(const float* __restrict__ fmap,          // (B*Ck, K) f32
                   const unsigned short* __restrict__ prfb, // (V, Kpad) bf16
                   const float* __restrict__ w2,            // offset: (+c_off*V)
                   float* __restrict__ partials,            // offset: (+tile_off*V)
                   int Ck, int K, int Kpad)
{
    const int tile  = blockIdx.x;
    const int mbase = tile * 16;
    const int lane  = threadIdx.x & 31;
    const int wv    = threadIdx.x >> 5;
    const int lo    = lane & 15;      // A: M row / B,D: N column
    const int hi    = lane >> 4;      // K-half selector

    const float* arow = fmap + (size_t)(mbase + lo) * K;
    // Full-vector region: rows 16B aligned (K%4==0) and whole chunk in-bounds.
    const int kvec = ((K & 3) == 0) ? (K & ~31) : 0;

    v8f_t acc[4] = {};  // four 16x16 f32 accumulator tiles

    // ---- main loop: no bounds checks, pure b128 loads ----
    int kk = 0;
    for (; kk < kvec; kk += 32) {
        // A fragment: 16x32 bf16 tile, contiguous k order (same as B).
        // lane L: M = L&15 ; K = base+i, base = 16*(L>>4)
        const int kA = kk + hi * 16;
        float a0[8], a1[8];
        load8v(arow, kA,      a0);
        load8v(arow, kA + 8,  a1);
        v16bf_t afrag;
#pragma unroll
        for (int i = 0; i < 8; ++i) {
            afrag[i]     = (__bf16)a0[i];
            afrag[i + 8] = (__bf16)a1[i];
        }
        // B fragment: lane L: N = L&15 ; K = (L>>4)*16 + i -> 32B contiguous.
        const int kB = kk + hi * 16;
#pragma unroll
        for (int t = 0; t < 4; ++t) {
            const int n = wv * 64 + t * 16 + lo;
            const v16bf_t bfrag = *reinterpret_cast<const v16bf_t*>(
                prfb + (size_t)n * Kpad + kB);
            acc[t] = __builtin_amdgcn_wmma_f32_16x16x32_bf16(
                false, afrag, false, bfrag, (short)0, acc[t], false, false);
        }
    }

    // ---- tail loop: branchless bounded scalar loads, zero-filled past K ----
    for (; kk < Kpad; kk += 32) {
        const int kA = kk + hi * 16;
        float a0[8], a1[8];
        load8t(arow, kA,      K, a0);
        load8t(arow, kA + 8,  K, a1);
        v16bf_t afrag;
#pragma unroll
        for (int i = 0; i < 8; ++i) {
            afrag[i]     = (__bf16)a0[i];
            afrag[i + 8] = (__bf16)a1[i];
        }
        const int kB = kk + hi * 16;
#pragma unroll
        for (int t = 0; t < 4; ++t) {
            const int n = wv * 64 + t * 16 + lo;
            const v16bf_t bfrag = *reinterpret_cast<const v16bf_t*>(
                prfb + (size_t)n * Kpad + kB);
            acc[t] = __builtin_amdgcn_wmma_f32_16x16x32_bf16(
                false, afrag, false, bfrag, (short)0, acc[t], false, false);
        }
    }

    // ---- epilogue: fold readout weights, reduce over the 16 c-rows ----
    // acc[t][j] holds pooled value for M = j + hi*8, N = wv*64 + t*16 + lo.
    const int cbase = mbase % Ck;
#pragma unroll
    for (int t = 0; t < 4; ++t) {
        const int n = wv * 64 + t * 16 + lo;
        float partial = 0.0f;
#pragma unroll
        for (int j = 0; j < 8; ++j) {
            const int c = cbase + hi * 8 + j;
            partial += acc[t][j] * w2[(size_t)c * NV + n];
        }
        // lanes L and L+16 cover c rows [cbase,cbase+8) and [cbase+8,cbase+16)
        partial += __shfl_xor(partial, 16, 32);
        if (hi == 0)
            partials[(size_t)tile * NV + n] = partial;
    }
}

// ---------------- final reduction -----------------------------------------

__global__ void reduce_kernel(const float* __restrict__ partials, // (NTILES, V)
                              const float* __restrict__ bias_eff, // (V)
                              float* __restrict__ out)            // (B, V)
{
    int idx = blockIdx.x * blockDim.x + threadIdx.x;
    if (idx >= NB * NV) return;
    int b = idx / NV;
    int v = idx - b * NV;
    const int toff[4] = {0, 512, 1536, 3584};
    const int ntls[4] = {4, 8, 16, 32};      // Ck/16 tiles per b
    float s = bias_eff[v];
#pragma unroll
    for (int k = 0; k < 4; ++k) {
        const int base = toff[k] + b * ntls[k];
        for (int i = 0; i < ntls[k]; ++i)
            s += partials[(size_t)(base + i) * NV + v];
    }
    out[idx] = s;
}

// ---------------- host launcher -------------------------------------------

extern "C" void kernel_launch(void* const* d_in, const int* in_sizes, int n_in,
                              void* d_out, int out_size, void* d_ws, size_t ws_size,
                              hipStream_t stream) {
    (void)in_sizes; (void)n_in; (void)out_size; (void)ws_size;

    // setup_inputs() dict order: fmap0,prf0,fmap1,prf1,fmap2,prf2,fmap3,prf3,
    //                            weights,bias,features_m,features_s
    const float* fmap[4] = {(const float*)d_in[0], (const float*)d_in[2],
                            (const float*)d_in[4], (const float*)d_in[6]};
    const float* prf[4]  = {(const float*)d_in[1], (const float*)d_in[3],
                            (const float*)d_in[5], (const float*)d_in[7]};
    const float* weights = (const float*)d_in[8];
    const float* bias    = (const float*)d_in[9];
    const float* fm      = (const float*)d_in[10];
    const float* fs      = (const float*)d_in[11];
    float* out = (float*)d_out;

    const int CH[4] = {64, 128, 256, 512};
    const int KK[4] = {3136, 784, 196, 49};
    const int KP[4] = {3136, 800, 224, 64};     // K rounded up to 32
    const int TOFF[4] = {0, 512, 1536, 3584};   // tile offsets
    const int COFF[4] = {0, 64, 192, 448};      // channel offsets

    // workspace carve-up (~22 MB): w2 | bias_eff | partials | prf_bf16
    float* w2       = (float*)d_ws;                       // NC*NV
    float* bias_eff = w2 + (size_t)NC * NV;               // NV
    float* partials = bias_eff + NV;                      // NTILES*NV
    unsigned short* prfb = (unsigned short*)(partials + (size_t)NTILES * NV);
    size_t pboff[4];
    {
        size_t o = 0;
        for (int k = 0; k < 4; ++k) { pboff[k] = o; o += (size_t)NV * KP[k]; }
    }

    prep_w2_kernel<<<(NC * NV + 255) / 256, 256, 0, stream>>>(weights, fs, w2);
    prep_bias_kernel<<<(NV + 63) / 64, 64, 0, stream>>>(bias, fm, w2, bias_eff);
    for (int k = 0; k < 4; ++k) {
        int total = NV * KP[k];
        prf_to_bf16_kernel<<<(total + 255) / 256, 256, 0, stream>>>(
            prf[k], prfb + pboff[k], KK[k], KP[k]);
    }
    for (int k = 0; k < 4; ++k) {
        int ntile = NB * CH[k] / 16;
        pooled_gemm_kernel<<<ntile, 256, 0, stream>>>(
            fmap[k], prfb + pboff[k],
            w2 + (size_t)COFF[k] * NV,
            partials + (size_t)TOFF[k] * NV,
            CH[k], KK[k], KP[k]);
    }
    reduce_kernel<<<(NB * NV + 255) / 256, 256, 0, stream>>>(partials, bias_eff, out);
}
